// AttentionFlowLayer_91285234909931
// MI455X (gfx1250) — compile-verified, matrix operand order checked by reading, not run on hardware
//
#include <hip/hip_runtime.h>
#include <hip/hip_bf16.h>

#define NB 16
#define LCC 2048
#define LQQ 256
#define DD 256
#define ROWS 64          // context rows per workgroup
#define WAVES 4          // 4 waves x 16 rows
#define QPAD 8           // LDS row padding (bf16 elems); keeps rows 16B aligned
#define QSTRIDE (DD + QPAD)

typedef __attribute__((ext_vector_type(16))) __bf16 v16bf;
typedef __attribute__((ext_vector_type(8)))  __bf16 v8bf;
typedef __attribute__((ext_vector_type(8)))  float  v8f;
typedef __attribute__((ext_vector_type(4)))  float  v4f;

__device__ __forceinline__ v8f wmma_bf16(v16bf a, v16bf b, v8f c) {
  // D = A(16x32 bf16) x B(32x16 bf16) + C(16x16 f32)
  return __builtin_amdgcn_wmma_f32_16x16x32_bf16(
      /*neg_a=*/false, a, /*neg_b=*/false, b,
      /*c_mod=*/(short)0, c, /*reuse_a=*/false, /*reuse_b=*/false);
}

// 16-bit A/B fragment from a bf16 LDS row: two contiguous 16B runs at element
// offsets kb and kb+16 (kb already includes the per-half +8 K shift).
// All addresses are 16B aligned -> 2x ds_load_b128.
__device__ __forceinline__ v16bf frag_from_lds(const __bf16* __restrict__ row, int kb) {
  v8bf lo = *(const v8bf*)(row + kb);
  v8bf hi = *(const v8bf*)(row + kb + 16);
  return __builtin_shufflevector(lo, hi, 0, 1, 2, 3, 4, 5, 6, 7,
                                 8, 9, 10, 11, 12, 13, 14, 15);
}

// A fragment from a global f32 row scaled by w_m (8 of these per wave-tile).
__device__ __forceinline__ v16bf frag_from_f32_scaled(const float* __restrict__ row,
                                                      const float* __restrict__ w, int kb) {
  v4f r0 = *(const v4f*)(row + kb);
  v4f r1 = *(const v4f*)(row + kb + 4);
  v4f r2 = *(const v4f*)(row + kb + 16);
  v4f r3 = *(const v4f*)(row + kb + 20);
  v4f w0 = *(const v4f*)(w + kb);
  v4f w1 = *(const v4f*)(w + kb + 4);
  v4f w2 = *(const v4f*)(w + kb + 16);
  v4f w3 = *(const v4f*)(w + kb + 20);
  r0 *= w0; r1 *= w1; r2 *= w2; r3 *= w3;
  v16bf r;
#pragma unroll
  for (int i = 0; i < 4; ++i) {
    r[i]      = (__bf16)r0[i];
    r[4 + i]  = (__bf16)r1[i];
    r[8 + i]  = (__bf16)r2[i];
    r[12 + i] = (__bf16)r3[i];
  }
  return r;
}

// ---------------------------------------------------------------------------
// Kernel 1: qdot[n][q] = dot(query[n,q,:], w_q)
// ---------------------------------------------------------------------------
__global__ __launch_bounds__(LQQ) void qdot_kernel(const float* __restrict__ qry,
                                                   const float* __restrict__ W,
                                                   float* __restrict__ qdot_g) {
  int n = blockIdx.x;
  int q = threadIdx.x;
  const float* row = qry + ((size_t)n * LQQ + q) * DD;
  float s = 0.f;
#pragma unroll 8
  for (int d = 0; d < DD; ++d) s += row[d] * W[DD + d];
  qdot_g[n * LQQ + q] = s;
}

// ---------------------------------------------------------------------------
// Kernel 2: per (n, 64-row block): S = cdot + qdot + (C*w_m)Q^T, softmax_q,
//           c2q = P*Q ; writes G sections 0,1,2 and rowmax to workspace.
// q-tiling is PERMUTED: tile qt covers columns q = qt + 16*n, so a lane's
// GEMM1 B-row offset is qt*QSTRIDE*2 bytes from its private base -> always
// fits the 16-bit DS offset immediate (no address VALU in the hot loop).
// The GEMM1 k-loop is deliberately ROLLED to bound register pressure:
// live set = acc[16] (128 VGPRs) + 1 A frag + 1 B frag.
// ---------------------------------------------------------------------------
__global__ __launch_bounds__(WAVES * 32, 1)
void attn_main_kernel(const float* __restrict__ ctx, const float* __restrict__ qry,
                      const float* __restrict__ W, const float* __restrict__ qdot_g,
                      float* __restrict__ rowmax_g, float* __restrict__ G) {
  // ~300 KB static LDS: legal on CDNA5 (320 KB per workgroup).
  __shared__ __bf16 QR[LQQ][QSTRIDE];     // Q row-major bf16  : B for GEMM1
  __shared__ __bf16 QT[DD][QSTRIDE];      // Q transposed bf16 : B for GEMM2
  __shared__ __bf16 P [ROWS][QSTRIDE];    // softmax probs     : A for GEMM2
  __shared__ float  qdot_s[LQQ];
  __shared__ float  cdot_s[ROWS];
  __shared__ float  wm_s[DD];

  const int tid  = threadIdx.x;
  const int nblk = LCC / ROWS;
  const int n    = blockIdx.x / nblk;
  const int c0   = (blockIdx.x % nblk) * ROWS;

  const float* ctxn = ctx + (size_t)n * LCC * DD;
  const float* qryn = qry + (size_t)n * LQQ * DD;

  // Stage Q (bf16) into LDS: one coalesced read of Q, two LDS images.
  for (int e = tid; e < LQQ * DD; e += WAVES * 32) {
    int q = e >> 8;           // D == 256
    int d = e & (DD - 1);
    __bf16 v = (__bf16)qryn[(size_t)q * DD + d];
    QR[q][d] = v;
    QT[d][q] = v;
  }
  for (int q = tid; q < LQQ; q += WAVES * 32) qdot_s[q] = qdot_g[n * LQQ + q];
  for (int d = tid; d < DD; d += WAVES * 32)  wm_s[d]   = W[2 * DD + d];
  if (tid < ROWS) {
    const float* row = ctxn + (size_t)(c0 + tid) * DD;
    __builtin_prefetch(row, 0, 0);        // global_prefetch_b8
    float s = 0.f;
#pragma unroll 8
    for (int d = 0; d < DD; ++d) s += row[d] * W[d];
    cdot_s[tid] = s;
  }
  __syncthreads();

  const int wave  = tid >> 5;
  const int lane  = tid & 31;
  const int lrow  = lane & 15;          // row (A) / col (B,C) index within tile
  const int half  = lane >> 4;          // 0: lanes 0-15, 1: lanes 16-31
  const int khalf = half * 8;           // K shift per ISA 16-bit layout
  const int mrow  = half * 8;           // C/D layout: M = v + 8*half
  const int wr0   = wave * 16;          // wave's first row within block

  const float* Arow = ctxn + (size_t)(c0 + wr0 + lrow) * DD;
  const __bf16* QRbase = &QR[16 * lrow][0];  // lane's GEMM1 B base (q = qt + 16*lrow)

  // ---- GEMM1: S-tile (16 x 256) in 16 x v8f accumulators ----
  v8f acc[16];
  v8f zero = {};
#pragma unroll
  for (int qt = 0; qt < 16; ++qt) acc[qt] = zero;

#pragma unroll 1
  for (int k = 0; k < 8; ++k) {
    const int kb = k * 32 + khalf;
    v16bf a = frag_from_f32_scaled(Arow, wm_s, kb);
#pragma unroll
    for (int qt = 0; qt < 16; ++qt) {
      v16bf b = frag_from_lds(QRbase + qt * QSTRIDE, kb);
      acc[qt] = wmma_bf16(a, b, acc[qt]);
    }
  }

  // ---- bias add + row softmax (rows split across the two 16-lane halves) ----
  // Column q held by this lane in tile qt is q = qt + 16*lrow.
  float rmax[8], rsum[8];
#pragma unroll
  for (int v = 0; v < 8; ++v) rmax[v] = -3.0e38f;
#pragma unroll
  for (int qt = 0; qt < 16; ++qt) {
    const float qb = qdot_s[qt + 16 * lrow];
#pragma unroll
    for (int v = 0; v < 8; ++v) {
      float s = acc[qt][v] + cdot_s[wr0 + mrow + v] + qb;
      acc[qt][v] = s;
      rmax[v] = fmaxf(rmax[v], s);
    }
  }
#pragma unroll
  for (int v = 0; v < 8; ++v) {
#pragma unroll
    for (int off = 1; off < 16; off <<= 1)
      rmax[v] = fmaxf(rmax[v], __shfl_xor(rmax[v], off, 32));
    rsum[v] = 0.f;
  }
#pragma unroll
  for (int qt = 0; qt < 16; ++qt)
#pragma unroll
    for (int v = 0; v < 8; ++v) {
      float e = __expf(acc[qt][v] - rmax[v]);
      acc[qt][v] = e;
      rsum[v] += e;
    }
#pragma unroll
  for (int v = 0; v < 8; ++v)
#pragma unroll
    for (int off = 1; off < 16; off <<= 1)
      rsum[v] += __shfl_xor(rsum[v], off, 32);

  if (lrow == 0) {
#pragma unroll
    for (int v = 0; v < 8; ++v)
      rowmax_g[n * LCC + c0 + wr0 + mrow + v] = rmax[v];
  }

  // Normalize and park P (bf16) in LDS at true column q = qt + 16*lrow.
#pragma unroll
  for (int v = 0; v < 8; ++v) rsum[v] = 1.f / rsum[v];
#pragma unroll
  for (int qt = 0; qt < 16; ++qt)
#pragma unroll
    for (int v = 0; v < 8; ++v)
      P[wr0 + mrow + v][qt + 16 * lrow] = (__bf16)(acc[qt][v] * rsum[v]);
  __syncthreads();

  // ---- GEMM2: c2q-tile = P (16 x 256) x Q (256 x 256), plus epilogue ----
  // Natural d tiling (d = dt*16 + lrow) so the G stores stay coalesced.
  const __bf16* Aprow = &P[wr0 + lrow][0];
  float* Gn = G + (size_t)n * LCC * (4 * DD);

#pragma unroll 2
  for (int dt = 0; dt < 16; ++dt) {
    const __bf16* Brow = &QT[dt * 16 + lrow][0];
    v8f acc2 = zero;
    v16bf a = frag_from_lds(Aprow, khalf);
    v16bf b = frag_from_lds(Brow, khalf);
#pragma unroll
    for (int k = 0; k < 8; ++k) {
      v16bf an = a, bn = b;
      if (k < 7) {
        const int kb2 = (k + 1) * 32 + khalf;
        an = frag_from_lds(Aprow, kb2);
        bn = frag_from_lds(Brow, kb2);
      }
      acc2 = wmma_bf16(a, b, acc2);
      a = an; b = bn;
    }
    const int d = dt * 16 + lrow;
#pragma unroll
    for (int v = 0; v < 8; ++v) {
      const int c = c0 + wr0 + mrow + v;
      float cv = ctxn[(size_t)c * DD + d];
      float aq = acc2[v];
      float* gr = Gn + (size_t)c * (4 * DD);
      gr[d]          = cv;        // section 0: context
      gr[DD + d]     = aq;        // section 1: c2q
      gr[2 * DD + d] = cv * aq;   // section 2: context * c2q
    }
  }
}

// ---------------------------------------------------------------------------
// Kernel 3a: b = softmax_c(rowmax[n,:]); q2c[n,d] = sum_c b[c]*ctx[c,d] -> ws
// ---------------------------------------------------------------------------
__global__ __launch_bounds__(DD) void q2c_reduce_kernel(const float* __restrict__ ctx,
                                                        const float* __restrict__ rowmax_g,
                                                        float* __restrict__ q2c_g) {
  __shared__ float red[DD];
  __shared__ float b_s[LCC];
  const int n   = blockIdx.x;
  const int tid = threadIdx.x;
  const float* rm = rowmax_g + n * LCC;

  float m = -3.0e38f;
  for (int c = tid; c < LCC; c += DD) m = fmaxf(m, rm[c]);
  red[tid] = m;
  __syncthreads();
  for (int off = DD / 2; off > 0; off >>= 1) {
    if (tid < off) red[tid] = fmaxf(red[tid], red[tid + off]);
    __syncthreads();
  }
  const float M = red[0];
  __syncthreads();

  float s = 0.f;
  for (int c = tid; c < LCC; c += DD) {
    float e = __expf(rm[c] - M);
    b_s[c] = e;
    s += e;
  }
  red[tid] = s;
  __syncthreads();
  for (int off = DD / 2; off > 0; off >>= 1) {
    if (tid < off) red[tid] += red[tid + off];
    __syncthreads();
  }
  const float invZ = 1.f / red[0];
  __syncthreads();

  const float* ctxn = ctx + (size_t)n * LCC * DD;
  float acc = 0.f;
  for (int c = 0; c < LCC; ++c) acc += b_s[c] * ctxn[(size_t)c * DD + tid];
  q2c_g[n * DD + tid] = acc * invZ;
}

// ---------------------------------------------------------------------------
// Kernel 3b: G section 3 = ctx * q2c (broadcast over c); wide elementwise.
// ---------------------------------------------------------------------------
#define G3ROWS 8
__global__ __launch_bounds__(DD) void g3_kernel(const float* __restrict__ ctx,
                                                const float* __restrict__ q2c_g,
                                                float* __restrict__ G) {
  const int blk = blockIdx.x;
  const int n   = blk / (LCC / G3ROWS);
  const int c0  = (blk % (LCC / G3ROWS)) * G3ROWS;
  const int d   = threadIdx.x;
  const float w = q2c_g[n * DD + d];
  const float* ctxn = ctx + (size_t)n * LCC * DD;
  float* Gn = G + (size_t)n * LCC * (4 * DD);
#pragma unroll
  for (int r = 0; r < G3ROWS; ++r) {
    const int c = c0 + r;
    Gn[(size_t)c * (4 * DD) + 3 * DD + d] = ctxn[(size_t)c * DD + d] * w;
  }
}

// ---------------------------------------------------------------------------
extern "C" void kernel_launch(void* const* d_in, const int* in_sizes, int n_in,
                              void* d_out, int out_size, void* d_ws, size_t ws_size,
                              hipStream_t stream) {
  const float* ctx = (const float*)d_in[0];   // (16, 2048, 256) f32
  const float* qry = (const float*)d_in[1];   // (16, 256, 256) f32
  const float* W   = (const float*)d_in[2];   // (768,) f32
  float* G = (float*)d_out;                   // (16, 2048, 1024) f32

  float* qdot_g   = (float*)d_ws;             // NB*LQQ floats
  float* rowmax_g = qdot_g + NB * LQQ;        // NB*LCC floats
  float* q2c_g    = rowmax_g + NB * LCC;      // NB*DD floats

  qdot_kernel<<<NB, LQQ, 0, stream>>>(qry, W, qdot_g);
  attn_main_kernel<<<NB * (LCC / ROWS), WAVES * 32, 0, stream>>>(
      ctx, qry, W, qdot_g, rowmax_g, G);
  q2c_reduce_kernel<<<NB, DD, 0, stream>>>(ctx, rowmax_g, q2c_g);
  g3_kernel<<<NB * (LCC / G3ROWS), DD, 0, stream>>>(ctx, q2c_g, G);
}
